// BaseTopoLayer_66391604461753
// MI455X (gfx1250) — compile-verified, matrix operand order checked by reading, not run on hardware
//
#include <hip/hip_runtime.h>
#include <math.h>

// ---------------------------------------------------------------------------
// CDNA5 (gfx1250) WMMA implementation of the TargetDiff-style BaseTopoLayer.
// All GEMMs run on V_WMMA_F32_16X16X32_F16 (wave32, f32 accumulate).
// ---------------------------------------------------------------------------

typedef __attribute__((ext_vector_type(16))) _Float16 v16h;
typedef __attribute__((ext_vector_type(8)))  float    v8f;

#define WMMA(a, b, c) \
  __builtin_amdgcn_wmma_f32_16x16x32_f16(false, (a), false, (b), (short)0, (c), false, false)

// Load one 16-lane fragment slice of a 16xK row-major f16 tile for K-step k0.
// Per ISA 7.12.2 (16-bit A 16x32): lanes 0-15 hold K {0..7,16..23},
// lanes 16-31 hold K {8..15,24..31}. B uses the same pattern on a
// pre-transposed [N][K] weight (column per lane).
__device__ __forceinline__ v16h frag_k32(const _Float16* __restrict__ p, int koff) {
  v16h f;
#pragma unroll
  for (int i = 0; i < 8; ++i) { f[i] = p[koff + i]; f[8 + i] = p[16 + koff + i]; }
  return f;
}

__device__ __forceinline__ void atomicMaxFloat(float* addr, float val) {
  if (val >= 0.f) atomicMax((int*)addr, __float_as_int(val));
  else            atomicMin((unsigned int*)addr, __float_as_uint(val));
}

// ---------------------------------------------------------------------------
// Weight prep: f32 [K,128] row-major  ->  f16 [128][Kpad] (transposed, padded)
// ---------------------------------------------------------------------------
__global__ void k_prep_w(_Float16* __restrict__ dst, const float* __restrict__ src,
                         int K, int Kpad) {
  int i = blockIdx.x * blockDim.x + threadIdx.x;
  if (i >= 128 * Kpad) return;
  int n = i / Kpad, k = i - n * Kpad;
  dst[i] = (k < K) ? (_Float16)src[k * 128 + n] : (_Float16)0.f;
}

__global__ void k_init(float* __restrict__ mx, float* __restrict__ den,
                       float* __restrict__ oacc, int N) {
  int i = blockIdx.x * blockDim.x + threadIdx.x;
  if (i < N * 128) oacc[i] = 0.f;
  if (i < N * 16)  { mx[i] = -INFINITY; den[i] = 0.f; }
}

// ---------------------------------------------------------------------------
// Node MLP (q):  h[N,128] -> W1[128,128] -> LN -> ReLU -> W2[128,128] -> q
// One block = 16 rows, 8 waves, each wave owns a 16x16 output tile.
// ---------------------------------------------------------------------------
__global__ __launch_bounds__(256) void k_mlp_q(
    const float* __restrict__ h,
    const _Float16* __restrict__ W1T, const float* __restrict__ b1,
    const float* __restrict__ g, const float* __restrict__ beta,
    const _Float16* __restrict__ W2T, const float* __restrict__ b2,
    float* __restrict__ q, int N) {
  __shared__ _Float16 sA[16][128];
  __shared__ float    sH[16][128];
  __shared__ _Float16 sN[16][128];
  __shared__ float    sP1[16][16], sP2[16][16];
  __shared__ float    sMu[16], sRs[16];

  const int m0 = blockIdx.x * 16;
  const int tid = threadIdx.x;

  const int wave = tid >> 5, lane = tid & 31;
  const int frow = lane & 15, koff = (lane & 16) ? 8 : 0;
  const int nn = wave * 16 + (lane & 15);
  const int mb = (lane & 16) ? 8 : 0;

  // hide first weight fetch behind LDS staging
  __builtin_prefetch(W1T + nn * 128, 0, 3);
  __builtin_prefetch(W2T + nn * 128, 0, 3);

  for (int i = tid; i < 16 * 128; i += 256) {
    int r = i >> 7, c = i & 127;
    int row = m0 + r; if (row >= N) row = N - 1;
    sA[r][c] = (_Float16)h[row * 128 + c];
  }
  __syncthreads();

  v8f acc = {0.f, 0.f, 0.f, 0.f, 0.f, 0.f, 0.f, 0.f};
  for (int k0 = 0; k0 < 128; k0 += 32) {
    v16h a = frag_k32(&sA[frow][k0], koff);
    v16h b = frag_k32(W1T + nn * 128 + k0, koff);
    acc = WMMA(a, b, acc);
  }
#pragma unroll
  for (int r = 0; r < 8; ++r) sH[mb + r][nn] = acc[r] + b1[nn];
  __syncthreads();

  // parallel one-pass LN stats: 16 lanes per row, each sums 8 elements
  {
    int row = tid >> 4, sub = tid & 15;
    float s = 0.f, s2 = 0.f;
#pragma unroll
    for (int j = 0; j < 8; ++j) {
      float x = sH[row][sub * 8 + j];
      s += x; s2 += x * x;
    }
    sP1[row][sub] = s; sP2[row][sub] = s2;
  }
  __syncthreads();
  if (tid < 16) {
    float S = 0.f, S2 = 0.f;
#pragma unroll
    for (int j = 0; j < 16; ++j) { S += sP1[tid][j]; S2 += sP2[tid][j]; }
    float mu = S * (1.f / 128.f);
    float var = S2 * (1.f / 128.f) - mu * mu;
    sMu[tid] = mu; sRs[tid] = rsqrtf(fmaxf(var, 0.f) + 1e-5f);
  }
  __syncthreads();
  for (int i = tid; i < 16 * 128; i += 256) {
    int r = i >> 7, c = i & 127;
    float x = (sH[r][c] - sMu[r]) * sRs[r] * g[c] + beta[c];
    sN[r][c] = (_Float16)fmaxf(x, 0.f);
  }
  __syncthreads();

  v8f acc2 = {0.f, 0.f, 0.f, 0.f, 0.f, 0.f, 0.f, 0.f};
  for (int k0 = 0; k0 < 128; k0 += 32) {
    v16h a = frag_k32(&sN[frow][k0], koff);
    v16h b = frag_k32(W2T + nn * 128 + k0, koff);
    acc2 = WMMA(a, b, acc2);
  }
#pragma unroll
  for (int r = 0; r < 8; ++r) {
    int row = m0 + mb + r;
    if (row < N) q[row * 128 + nn] = acc2[r] + b2[nn];
  }
}

// ---------------------------------------------------------------------------
// Edge kernel: build kv_input [16][288] (f16, zero-padded), run xk & xv MLPs
// sharing the A-fragment, then per-(edge,head) scores = q[dst].k / sqrt(8).
// Stores v*(e_w) as f16 [E,128] and scores f32 [E,16].
// ---------------------------------------------------------------------------
__global__ __launch_bounds__(256) void k_edge(
    const float* __restrict__ h, const float* __restrict__ r_feat,
    const float* __restrict__ edge_feat, const float* __restrict__ e_w,
    const int* __restrict__ srcIdx, const int* __restrict__ dstIdx,
    const _Float16* __restrict__ Wk1T, const float* __restrict__ bk1,
    const float* __restrict__ gk, const float* __restrict__ betak,
    const _Float16* __restrict__ Wk2T, const float* __restrict__ bk2,
    const _Float16* __restrict__ Wv1T, const float* __restrict__ bv1,
    const float* __restrict__ gv, const float* __restrict__ betav,
    const _Float16* __restrict__ Wv2T, const float* __restrict__ bv2,
    const float* __restrict__ qbuf,
    _Float16* __restrict__ vbuf, float* __restrict__ sbuf, int E) {
  __shared__ _Float16 sKV[16][288];
  __shared__ float    sHk[16][128];
  __shared__ float    sHv[16][128];
  __shared__ _Float16 sNk[16][128];
  __shared__ _Float16 sNv[16][128];
  __shared__ float    sK[16][128];
  __shared__ float    sP1[32][8], sP2[32][8];
  __shared__ float    sMu[32], sRs[32], sEw[16];
  __shared__ int      sDst[16];

  const int e0 = blockIdx.x * 16;
  const int tid = threadIdx.x;

  const int wave = tid >> 5, lane = tid & 31;
  const int frow = lane & 15, koff = (lane & 16) ? 8 : 0;
  const int nn = wave * 16 + (lane & 15);
  const int mb = (lane & 16) ? 8 : 0;

  __builtin_prefetch(Wk1T + nn * 288, 0, 3);
  __builtin_prefetch(Wv1T + nn * 288, 0, 3);

  if (tid < 16) {
    int e = e0 + tid; if (e >= E) e = E - 1;
    sEw[tid] = e_w[e];
    sDst[tid] = dstIdx[e];
  }
  // gather kv_input: each thread owns one edge row, strides across columns
  {
    int r = tid >> 4, cs = tid & 15;
    int e = e0 + r; if (e >= E) e = E - 1;
    int di = dstIdx[e], si = srcIdx[e];
    for (int c = cs; c < 288; c += 16) {
      float x;
      if      (c < 4)   x = edge_feat[e * 4 + c];
      else if (c < 24)  x = r_feat[e * 20 + (c - 4)];
      else if (c < 152) x = h[di * 128 + (c - 24)];
      else if (c < 280) x = h[si * 128 + (c - 152)];
      else              x = 0.f;
      sKV[r][c] = (_Float16)x;
    }
  }
  __syncthreads();

  // stage 1 for k and v (shared A-frag)
  v8f ak = {0.f, 0.f, 0.f, 0.f, 0.f, 0.f, 0.f, 0.f};
  v8f av = {0.f, 0.f, 0.f, 0.f, 0.f, 0.f, 0.f, 0.f};
  for (int k0 = 0; k0 < 288; k0 += 32) {
    v16h a  = frag_k32(&sKV[frow][k0], koff);
    v16h bk = frag_k32(Wk1T + nn * 288 + k0, koff);
    v16h bv = frag_k32(Wv1T + nn * 288 + k0, koff);
    ak = WMMA(a, bk, ak);
    av = WMMA(a, bv, av);
  }
#pragma unroll
  for (int r = 0; r < 8; ++r) {
    sHk[mb + r][nn] = ak[r] + bk1[nn];
    sHv[mb + r][nn] = av[r] + bv1[nn];
  }
  __syncthreads();

  // parallel one-pass LN stats for both hiddens: 8 lanes per row x 32 rows
  {
    int row = tid >> 3, sub = tid & 7;
    const float* Hrow = (row < 16) ? &sHk[row][0] : &sHv[row - 16][0];
    float s = 0.f, s2 = 0.f;
#pragma unroll
    for (int j = 0; j < 16; ++j) {
      float x = Hrow[sub * 16 + j];
      s += x; s2 += x * x;
    }
    sP1[row][sub] = s; sP2[row][sub] = s2;
  }
  __syncthreads();
  if (tid < 32) {
    float S = 0.f, S2 = 0.f;
#pragma unroll
    for (int j = 0; j < 8; ++j) { S += sP1[tid][j]; S2 += sP2[tid][j]; }
    float mu = S * (1.f / 128.f);
    float var = S2 * (1.f / 128.f) - mu * mu;
    sMu[tid] = mu; sRs[tid] = rsqrtf(fmaxf(var, 0.f) + 1e-5f);
  }
  __syncthreads();
  for (int i = tid; i < 2 * 16 * 128; i += 256) {
    int which = i >> 11, r = (i >> 7) & 15, c = i & 127;
    if (which == 0) {
      float x = (sHk[r][c] - sMu[r]) * sRs[r] * gk[c] + betak[c];
      sNk[r][c] = (_Float16)fmaxf(x, 0.f);
    } else {
      float x = (sHv[r][c] - sMu[16 + r]) * sRs[16 + r] * gv[c] + betav[c];
      sNv[r][c] = (_Float16)fmaxf(x, 0.f);
    }
  }
  __syncthreads();

  // stage 2
  v8f ck = {0.f, 0.f, 0.f, 0.f, 0.f, 0.f, 0.f, 0.f};
  v8f cv = {0.f, 0.f, 0.f, 0.f, 0.f, 0.f, 0.f, 0.f};
  for (int k0 = 0; k0 < 128; k0 += 32) {
    v16h aKk = frag_k32(&sNk[frow][k0], koff);
    v16h bKk = frag_k32(Wk2T + nn * 128 + k0, koff);
    ck = WMMA(aKk, bKk, ck);
    v16h aVv = frag_k32(&sNv[frow][k0], koff);
    v16h bVv = frag_k32(Wv2T + nn * 128 + k0, koff);
    cv = WMMA(aVv, bVv, cv);
  }
#pragma unroll
  for (int r = 0; r < 8; ++r) {
    int m = mb + r, e = e0 + m;
    sK[m][nn] = ck[r] + bk2[nn];
    if (e < E) vbuf[e * 128 + nn] = (_Float16)((cv[r] + bv2[nn]) * sEw[m]);
  }
  __syncthreads();

  // per-(edge,head) attention scores
  {
    int el = tid >> 4, hh = tid & 15;
    const float* qrow = qbuf + sDst[el] * 128 + hh * 8;
    float s = 0.f;
#pragma unroll
    for (int d = 0; d < 8; ++d) s += qrow[d] * sK[el][hh * 8 + d];
    int e = e0 + el;
    if (e < E) sbuf[e * 16 + hh] = s * 0.3535533906f;  // 1/sqrt(8)
  }
}

// ---------------------------------------------------------------------------
// Segment softmax over edges sharing dst, then alpha*v scatter-add.
// ---------------------------------------------------------------------------
__global__ void k_seg_max(const float* __restrict__ sbuf, const int* __restrict__ dstIdx,
                          float* __restrict__ mx, int E16) {
  int i = blockIdx.x * blockDim.x + threadIdx.x;
  if (i >= E16) return;
  int e = i >> 4, hh = i & 15;
  atomicMaxFloat(&mx[dstIdx[e] * 16 + hh], sbuf[i]);
}

__global__ void k_seg_exp(float* __restrict__ sbuf, const int* __restrict__ dstIdx,
                          const float* __restrict__ mx, float* __restrict__ den, int E16) {
  int i = blockIdx.x * blockDim.x + threadIdx.x;
  if (i >= E16) return;
  int e = i >> 4, hh = i & 15;
  int d = dstIdx[e];
  float ex = __expf(sbuf[i] - mx[d * 16 + hh]);
  sbuf[i] = ex;
  atomicAdd(&den[d * 16 + hh], ex);
}

__global__ void k_scatter(const float* __restrict__ sbuf, const float* __restrict__ den,
                          const _Float16* __restrict__ vbuf, const int* __restrict__ dstIdx,
                          float* __restrict__ oacc, int E128) {
  int i = blockIdx.x * blockDim.x + threadIdx.x;
  if (i >= E128) return;
  int e = i >> 7, c = i & 127, hh = c >> 3;
  int d = dstIdx[e];
  float alpha = sbuf[e * 16 + hh] / den[d * 16 + hh];
  atomicAdd(&oacc[d * 128 + c], alpha * (float)vbuf[i]);
}

// ---------------------------------------------------------------------------
// Output MLP: concat(attn_out, h) [N,256] -> 128 -> LN -> ReLU -> 128
// ---------------------------------------------------------------------------
__global__ __launch_bounds__(256) void k_mlp_out(
    const float* __restrict__ oacc, const float* __restrict__ h,
    const _Float16* __restrict__ W1T, const float* __restrict__ b1,
    const float* __restrict__ g, const float* __restrict__ beta,
    const _Float16* __restrict__ W2T, const float* __restrict__ b2,
    float* __restrict__ out, int N) {
  __shared__ _Float16 sX[16][256];
  __shared__ float    sH[16][128];
  __shared__ _Float16 sN[16][128];
  __shared__ float    sP1[16][16], sP2[16][16];
  __shared__ float    sMu[16], sRs[16];

  const int m0 = blockIdx.x * 16;
  const int tid = threadIdx.x;

  const int wave = tid >> 5, lane = tid & 31;
  const int frow = lane & 15, koff = (lane & 16) ? 8 : 0;
  const int nn = wave * 16 + (lane & 15);
  const int mb = (lane & 16) ? 8 : 0;

  __builtin_prefetch(W1T + nn * 256, 0, 3);
  __builtin_prefetch(W2T + nn * 128, 0, 3);

  for (int i = tid; i < 16 * 256; i += 256) {
    int r = i >> 8, c = i & 255;
    int row = m0 + r; if (row >= N) row = N - 1;
    float x = (c < 128) ? oacc[row * 128 + c] : h[row * 128 + (c - 128)];
    sX[r][c] = (_Float16)x;
  }
  __syncthreads();

  v8f acc = {0.f, 0.f, 0.f, 0.f, 0.f, 0.f, 0.f, 0.f};
  for (int k0 = 0; k0 < 256; k0 += 32) {
    v16h a = frag_k32(&sX[frow][k0], koff);
    v16h b = frag_k32(W1T + nn * 256 + k0, koff);
    acc = WMMA(a, b, acc);
  }
#pragma unroll
  for (int r = 0; r < 8; ++r) sH[mb + r][nn] = acc[r] + b1[nn];
  __syncthreads();

  {
    int row = tid >> 4, sub = tid & 15;
    float s = 0.f, s2 = 0.f;
#pragma unroll
    for (int j = 0; j < 8; ++j) {
      float x = sH[row][sub * 8 + j];
      s += x; s2 += x * x;
    }
    sP1[row][sub] = s; sP2[row][sub] = s2;
  }
  __syncthreads();
  if (tid < 16) {
    float S = 0.f, S2 = 0.f;
#pragma unroll
    for (int j = 0; j < 16; ++j) { S += sP1[tid][j]; S2 += sP2[tid][j]; }
    float mu = S * (1.f / 128.f);
    float var = S2 * (1.f / 128.f) - mu * mu;
    sMu[tid] = mu; sRs[tid] = rsqrtf(fmaxf(var, 0.f) + 1e-5f);
  }
  __syncthreads();
  for (int i = tid; i < 16 * 128; i += 256) {
    int r = i >> 7, c = i & 127;
    float x = (sH[r][c] - sMu[r]) * sRs[r] * g[c] + beta[c];
    sN[r][c] = (_Float16)fmaxf(x, 0.f);
  }
  __syncthreads();

  v8f acc2 = {0.f, 0.f, 0.f, 0.f, 0.f, 0.f, 0.f, 0.f};
  for (int k0 = 0; k0 < 128; k0 += 32) {
    v16h a = frag_k32(&sN[frow][k0], koff);
    v16h b = frag_k32(W2T + nn * 128 + k0, koff);
    acc2 = WMMA(a, b, acc2);
  }
#pragma unroll
  for (int r = 0; r < 8; ++r) {
    int row = m0 + mb + r;
    if (row < N) out[row * 128 + nn] = acc2[r] + b2[nn];
  }
}

// ---------------------------------------------------------------------------
// Host launcher
// ---------------------------------------------------------------------------
extern "C" void kernel_launch(void* const* d_in, const int* in_sizes, int n_in,
                              void* d_out, int out_size, void* d_ws, size_t ws_size,
                              hipStream_t stream) {
  const float* h         = (const float*)d_in[0];
  const float* r_feat    = (const float*)d_in[1];
  const float* edge_feat = (const float*)d_in[2];
  const int*   edge_idx  = (const int*)d_in[3];
  const float* e_w       = (const float*)d_in[4];

  const float* xk_W1 = (const float*)d_in[5];  const float* xk_b1 = (const float*)d_in[6];
  const float* xk_g  = (const float*)d_in[7];  const float* xk_be = (const float*)d_in[8];
  const float* xk_W2 = (const float*)d_in[9];  const float* xk_b2 = (const float*)d_in[10];
  const float* xv_W1 = (const float*)d_in[11]; const float* xv_b1 = (const float*)d_in[12];
  const float* xv_g  = (const float*)d_in[13]; const float* xv_be = (const float*)d_in[14];
  const float* xv_W2 = (const float*)d_in[15]; const float* xv_b2 = (const float*)d_in[16];
  const float* xq_W1 = (const float*)d_in[17]; const float* xq_b1 = (const float*)d_in[18];
  const float* xq_g  = (const float*)d_in[19]; const float* xq_be = (const float*)d_in[20];
  const float* xq_W2 = (const float*)d_in[21]; const float* xq_b2 = (const float*)d_in[22];
  const float* ou_W1 = (const float*)d_in[23]; const float* ou_b1 = (const float*)d_in[24];
  const float* ou_g  = (const float*)d_in[25]; const float* ou_be = (const float*)d_in[26];
  const float* ou_W2 = (const float*)d_in[27]; const float* ou_b2 = (const float*)d_in[28];

  const int N = in_sizes[0] / 128;
  const int E = in_sizes[4];
  const int* srcIdx = edge_idx;
  const int* dstIdx = edge_idx + E;

  // workspace carve-out (256B aligned)
  char* base = (char*)d_ws;
  size_t off = 0;
  auto carve = [&](size_t bytes) -> char* {
    char* p = base + off;
    off = (off + bytes + 255) & ~(size_t)255;
    return p;
  };
  _Float16* wk1T = (_Float16*)carve((size_t)128 * 288 * 2);
  _Float16* wv1T = (_Float16*)carve((size_t)128 * 288 * 2);
  _Float16* wq1T = (_Float16*)carve((size_t)128 * 128 * 2);
  _Float16* wo1T = (_Float16*)carve((size_t)128 * 256 * 2);
  _Float16* wk2T = (_Float16*)carve((size_t)128 * 128 * 2);
  _Float16* wv2T = (_Float16*)carve((size_t)128 * 128 * 2);
  _Float16* wq2T = (_Float16*)carve((size_t)128 * 128 * 2);
  _Float16* wo2T = (_Float16*)carve((size_t)128 * 128 * 2);
  float*    qbuf = (float*)carve((size_t)N * 128 * 4);
  _Float16* vbuf = (_Float16*)carve((size_t)E * 128 * 2);
  float*    sbuf = (float*)carve((size_t)E * 16 * 4);
  float*    mx   = (float*)carve((size_t)N * 16 * 4);
  float*    den  = (float*)carve((size_t)N * 16 * 4);
  float*    oacc = (float*)carve((size_t)N * 128 * 4);

  const int T = 256;
  // 1. weight prep
  k_prep_w<<<(128 * 288 + T - 1) / T, T, 0, stream>>>(wk1T, xk_W1, 280, 288);
  k_prep_w<<<(128 * 288 + T - 1) / T, T, 0, stream>>>(wv1T, xv_W1, 280, 288);
  k_prep_w<<<(128 * 128 + T - 1) / T, T, 0, stream>>>(wq1T, xq_W1, 128, 128);
  k_prep_w<<<(128 * 256 + T - 1) / T, T, 0, stream>>>(wo1T, ou_W1, 256, 256);
  k_prep_w<<<(128 * 128 + T - 1) / T, T, 0, stream>>>(wk2T, xk_W2, 128, 128);
  k_prep_w<<<(128 * 128 + T - 1) / T, T, 0, stream>>>(wv2T, xv_W2, 128, 128);
  k_prep_w<<<(128 * 128 + T - 1) / T, T, 0, stream>>>(wq2T, xq_W2, 128, 128);
  k_prep_w<<<(128 * 128 + T - 1) / T, T, 0, stream>>>(wo2T, ou_W2, 128, 128);
  // 2. init accumulators
  k_init<<<((size_t)N * 128 + T - 1) / T, T, 0, stream>>>(mx, den, oacc, N);
  // 3. q = MLP_xq(h)
  k_mlp_q<<<(N + 15) / 16, T, 0, stream>>>(h, wq1T, xq_b1, xq_g, xq_be, wq2T, xq_b2, qbuf, N);
  // 4. edge MLPs + scores
  k_edge<<<(E + 15) / 16, T, 0, stream>>>(h, r_feat, edge_feat, e_w, srcIdx, dstIdx,
                                          wk1T, xk_b1, xk_g, xk_be, wk2T, xk_b2,
                                          wv1T, xv_b1, xv_g, xv_be, wv2T, xv_b2,
                                          qbuf, vbuf, sbuf, E);
  // 5. segment softmax + scatter
  k_seg_max<<<((size_t)E * 16 + T - 1) / T, T, 0, stream>>>(sbuf, dstIdx, mx, E * 16);
  k_seg_exp<<<((size_t)E * 16 + T - 1) / T, T, 0, stream>>>(sbuf, dstIdx, mx, den, E * 16);
  k_scatter<<<((size_t)E * 128 + T - 1) / T, T, 0, stream>>>(sbuf, den, vbuf, dstIdx, oacc,
                                                             E * 128);
  // 6. final MLP on concat(attn_out, h)
  k_mlp_out<<<(N + 15) / 16, T, 0, stream>>>(oacc, h, wo1T, ou_b1, ou_g, ou_be, wo2T, ou_b2,
                                             (float*)d_out, N);
}
